// GroupedQueryAttention_43241730736392
// MI455X (gfx1250) — matrix operand order repaired and checked by reading, not compiled, not run on hardware
//
#include <hip/hip_runtime.h>
#include <hip/hip_bf16.h>

// ---------------------------------------------------------------------------
// Types & helpers
// ---------------------------------------------------------------------------
typedef __attribute__((ext_vector_type(16))) __bf16 bf16x16;
typedef __attribute__((ext_vector_type(8)))  float  f32x8;
typedef __attribute__((ext_vector_type(4)))  unsigned int u32x4;
typedef __attribute__((ext_vector_type(8)))  int    i32x8;
typedef __attribute__((ext_vector_type(4)))  int    i32x4;

// TDM builtin availability; this toolchain exposes the 6-arg form
// (uint32x4 g0, int32x8 g1, int32x4 g2, int32x4 g3, int32x8 g4, i32 cpol).
#if defined(__has_builtin)
#  if __has_builtin(__builtin_amdgcn_tensor_load_to_lds) && \
      __has_builtin(__builtin_amdgcn_s_wait_tensorcnt)
#    define USE_TDM 1
#  endif
#endif
#ifndef USE_TDM
#  define USE_TDM 0
#endif

__device__ __forceinline__ unsigned short f32_to_bf16(float f) {
    unsigned int u = __builtin_bit_cast(unsigned int, f);
    u += 0x7fffu + ((u >> 16) & 1u);          // round-to-nearest-even
    return (unsigned short)(u >> 16);
}
__device__ __forceinline__ __bf16 bits_to_bf16(unsigned short h) {
    return __builtin_bit_cast(__bf16, h);
}

__device__ __forceinline__ f32x8 wmma_bf16(bf16x16 a, bf16x16 b, f32x8 c) {
    // D = A(16x32 bf16) * B(32x16 bf16) + C(16x16 f32)
    return __builtin_amdgcn_wmma_f32_16x16x32_bf16(
        /*neg_a=*/false, a, /*neg_b=*/false, b,
        /*c_mod=*/(short)0, c, /*reuse_a=*/false, /*reuse_b=*/false);
}

// A-fragment (16x32, 16-bit): lane L<16 -> row M=L, K = e (e<8) / 16+(e-8);
// lane L>=16 -> row M=L-16, K = 8+e / 24+(e-8).  (ISA 7.12.2)
__device__ __forceinline__ bf16x16 load_a_frag(const unsigned short* base,
                                               int stride, int rowBase,
                                               int kBase, int lane) {
    const unsigned short* p =
        base + (size_t)(rowBase + (lane & 15)) * stride + kBase;
    int kh = ((lane >> 4) & 1) * 8;
    bf16x16 a;
#pragma unroll
    for (int e = 0; e < 8; ++e) a[e] = bits_to_bf16(p[kh + e]);
#pragma unroll
    for (int e = 0; e < 8; ++e) a[8 + e] = bits_to_bf16(p[16 + kh + e]);
    return a;
}

// B-fragment (32x16, 16-bit): lane L -> col N=L%16, K = (L/16)*8 + e (e<8) / 16+(L/16)*8+(e-8).
// Memory is stored "per column": base[(col)*stride + k], k contiguous.
__device__ __forceinline__ bf16x16 load_b_frag(const unsigned short* base,
                                               int stride, int colBase,
                                               int kBase, int lane) {
    const unsigned short* p = base + (size_t)(colBase + (lane & 15)) * stride +
                              kBase;
    int kh = ((lane >> 4) & 1) * 8;
    bf16x16 b;
#pragma unroll
    for (int e = 0; e < 8; ++e) b[e] = bits_to_bf16(p[kh + e]);
#pragma unroll
    for (int e = 0; e < 8; ++e) b[8 + e] = bits_to_bf16(p[16 + kh + e]);
    return b;
}

// ---------------------------------------------------------------------------
// GEMM: C[M,N] = A[M,K] * W[N,K]^T + bias[N]
// A is f32 or bf16(ushort); C is f32 or bf16(ushort).
// 128 threads (4 waves), 64x64 output tile, K stepped by 64 (8 WMMA/wave/iter).
// M,N multiples of 64; K multiple of 64 (true for all calls: K = 1024).
// ---------------------------------------------------------------------------
template <bool A_BF16, bool OUT_BF16>
__global__ __launch_bounds__(128) void gemm_bt_kernel(const void* __restrict__ A_,
                                                      const float* __restrict__ W,
                                                      const float* __restrict__ bias,
                                                      void* __restrict__ C_,
                                                      int M, int N, int K) {
    __shared__ unsigned short As[64 * 64];   // 8 KB
    __shared__ unsigned short Ws[64 * 64];   // 8 KB
    const int tid = threadIdx.x;
    const int lane = tid & 31, wave = tid >> 5;
    const int rowTile = blockIdx.x * 64;
    const int colTile = blockIdx.y * 64;
    const int wr = (wave >> 1) * 32;   // wave's row quadrant
    const int wc = (wave & 1) * 32;    // wave's col quadrant

    f32x8 acc[2][2] = {};

    for (int k0 = 0; k0 < K; k0 += 64) {
        __syncthreads();
        // Prefetch next K-slice into GL2 (global_prefetch_b8).
        if (k0 + 64 < K) {
            int pr = tid >> 1, pk = (tid & 1) * 32;
            if constexpr (A_BF16)
                __builtin_prefetch(((const unsigned short*)A_) +
                                       (size_t)(rowTile + pr) * K + k0 + 64 + pk,
                                   0, 1);
            else
                __builtin_prefetch(((const float*)A_) +
                                       (size_t)(rowTile + pr) * K + k0 + 64 + pk,
                                   0, 1);
            __builtin_prefetch(W + (size_t)(colTile + pr) * K + k0 + 64 + pk, 0, 1);
        }
#pragma unroll
        for (int e = 0; e < 32; ++e) {
            int idx = tid * 32 + e;            // 0..4095 -> 64 rows x 64 k
            int r = idx >> 6, kk = idx & 63;
            unsigned short av;
            if constexpr (A_BF16)
                av = ((const unsigned short*)A_)[(size_t)(rowTile + r) * K + k0 + kk];
            else
                av = f32_to_bf16(((const float*)A_)[(size_t)(rowTile + r) * K + k0 + kk]);
            As[r * 64 + kk] = av;
            Ws[r * 64 + kk] =
                f32_to_bf16(W[(size_t)(colTile + r) * K + k0 + kk]);
        }
        __syncthreads();

#pragma unroll
        for (int ks = 0; ks < 2; ++ks) {
            bf16x16 a0 = load_a_frag(As, 64, wr, ks * 32, lane);
            bf16x16 a1 = load_a_frag(As, 64, wr + 16, ks * 32, lane);
            bf16x16 b0 = load_b_frag(Ws, 64, wc, ks * 32, lane);
            bf16x16 b1 = load_b_frag(Ws, 64, wc + 16, ks * 32, lane);
            acc[0][0] = wmma_bf16(a0, b0, acc[0][0]);
            acc[0][1] = wmma_bf16(a0, b1, acc[0][1]);
            acc[1][0] = wmma_bf16(a1, b0, acc[1][0]);
            acc[1][1] = wmma_bf16(a1, b1, acc[1][1]);
        }
    }

    // C/D layout: VGPR r -> M = r (lanes 0-15) or r+8 (lanes 16-31); N = lane%16
    const int nloc = lane & 15;
    const int mOff = (lane >> 4) * 8;
#pragma unroll
    for (int i = 0; i < 2; ++i)
#pragma unroll
        for (int j = 0; j < 2; ++j)
#pragma unroll
            for (int r = 0; r < 8; ++r) {
                int m = rowTile + wr + i * 16 + mOff + r;
                int n = colTile + wc + j * 16 + nloc;
                float v = acc[i][j][r] + bias[n];
                if constexpr (OUT_BF16)
                    ((unsigned short*)C_)[(size_t)m * N + n] = f32_to_bf16(v);
                else
                    ((float*)C_)[(size_t)m * N + n] = v;
            }
}

// ---------------------------------------------------------------------------
// RoPE: x1=x[::2], x2=x[1::2]; out = [x1*c - x2*s, x1*s + x2*c]  (head dim 64)
// Reads f32 projections, writes bf16.
// ---------------------------------------------------------------------------
__global__ __launch_bounds__(256) void rope_kernel(const float* __restrict__ lin,
                                                   unsigned short* __restrict__ out,
                                                   int heads, int L, int total) {
    int t = blockIdx.x * blockDim.x + threadIdx.x;
    if (t >= total) return;
    int i  = t & 31;                 // rotary pair index 0..31
    int h  = (t >> 5) % heads;
    int bl = t / (32 * heads);       // 0..B*L-1
    int l  = bl % L;

    float inv = powf(10000.0f, -(float)(2 * i) / 64.0f);
    float ang = (float)l * inv;
    float c = cosf(ang), s = sinf(ang);

    size_t base = ((size_t)bl * heads + h) * 64;
    float x1 = lin[base + 2 * i];
    float x2 = lin[base + 2 * i + 1];
    out[base + i]      = f32_to_bf16(x1 * c - x2 * s);
    out[base + 32 + i] = f32_to_bf16(x1 * s + x2 * c);
}

// ---------------------------------------------------------------------------
// Flash-style GQA attention. One block = (b, h, 64 query rows); 4 waves,
// each wave owns 16 query rows. Keys streamed 32 at a time through LDS.
// The K chunk is a pure 2D strided copy -> Tensor Data Mover (TDM);
// V needs a transpose so it stays on the manual path.
// q,k layouts: [B,L,H,64] / [B,L,G,64] bf16. out: [B,L,H*64] bf16.
// ---------------------------------------------------------------------------
__global__ __launch_bounds__(128) void attn_kernel(const unsigned short* __restrict__ qb_,
                                                   const unsigned short* __restrict__ kb_,
                                                   const unsigned short* __restrict__ vb_,
                                                   unsigned short* __restrict__ out,
                                                   int B, int L, int H, int G) {
    __shared__ unsigned short Ks[32 * 64];    // [m][d]  -> B-frags for Q*K^T
    __shared__ unsigned short Vt[64 * 32];    // [d][m]  -> B-frags for P*V
    __shared__ unsigned short Ps[4][16 * 32]; // per-wave P transpose scratch

    const int tid = threadIdx.x, lane = tid & 31, wave = tid >> 5;
    const int qtiles = L / 64;
    const int qb = blockIdx.x % qtiles;
    const int bh = blockIdx.x / qtiles;
    const int h = bh % H, b = bh / H;
    const int g = h / (H / G);
    const int qRow0 = qb * 64 + wave * 16;
    const int nloc = lane & 15;
    const int mOff = (lane >> 4) * 8;
    const float scale = 0.125f;   // 1/sqrt(64)

    // Q A-fragments straight from global (row stride = H*64)
    const unsigned short* qbase =
        qb_ + (((size_t)b * L + qRow0) * (size_t)H + h) * 64;
    bf16x16 qa0 = load_a_frag(qbase, H * 64, 0, 0, lane);
    bf16x16 qa1 = load_a_frag(qbase, H * 64, 0, 32, lane);

#if USE_TDM
    // Flat LDS pointers carry the LDS byte offset in their low 32 bits.
    const unsigned ldsKs = (unsigned)(unsigned long long)(const void*)Ks;
#endif

    f32x8 O[4] = {};
    float m_run[8], l_run[8];
#pragma unroll
    for (int r = 0; r < 8; ++r) { m_run[r] = -__builtin_inff(); l_run[r] = 0.0f; }

    for (int m0 = 0; m0 < L; m0 += 32) {
        __syncthreads();
#if USE_TDM
        if (wave == 0) {
            // D# per CDNA5 ISA: 2D tile, 2-byte elems,
            // tile 64(x) x 32(y), row stride G*64 elems, dest = Ks.
            unsigned long long ga = (unsigned long long)(const void*)
                (kb_ + (((size_t)b * L + m0) * G + g) * 64);
            u32x4 g0;
            g0[0] = 1u;                                   // count=1 (valid D#)
            g0[1] = ldsKs;                                // lds_addr
            g0[2] = (unsigned)(ga & 0xffffffffull);       // global_addr[31:0]
            g0[3] = (unsigned)((ga >> 32) & 0x1ffffffull) // global_addr[56:32]
                    | (2u << 30);                         // type=2 ("image")
            i32x8 g1;
            g1[0] = 0x00010000;         // data_size=1 -> 2 bytes
            g1[1] = (int)(64u << 16);   // tensor_dim0 = 64  (bits 79:48)
            g1[2] = (int)(32u << 16);   // tensor_dim1 = 32  (bits 111:80)
            g1[3] = (int)(64u << 16);   // tile_dim0   = 64  (bits 127:112)
            g1[4] = 32;                 // tile_dim1   = 32  (bits 143:128)
            g1[5] = 256;                // tensor_dim0_stride = G*64 elems
            g1[6] = 0;
            g1[7] = 0;
            i32x4 gz  = {0, 0, 0, 0};            // groups 2/3 unused (2D)
            i32x8 gz8 = {0, 0, 0, 0, 0, 0, 0, 0};
            __builtin_amdgcn_tensor_load_to_lds(g0, g1, gz, gz, gz8, 0);
            __builtin_amdgcn_s_wait_tensorcnt(0);
        }
        // V chunk (manual: needs [d][m] transpose)
#pragma unroll
        for (int e = 0; e < 16; ++e) {
            int idx = tid * 16 + e;            // 0..2047 -> 32 m x 64 d
            int mm = idx >> 6, d = idx & 63;
            size_t gk = (((size_t)b * L + m0 + mm) * G + g) * 64 + d;
            Vt[d * 32 + mm] = vb_[gk];
        }
#else
#pragma unroll
        for (int e = 0; e < 16; ++e) {
            int idx = tid * 16 + e;            // 0..2047 -> 32 m x 64 d
            int mm = idx >> 6, d = idx & 63;
            size_t gk = (((size_t)b * L + m0 + mm) * G + g) * 64 + d;
            Ks[mm * 64 + d] = kb_[gk];
            Vt[d * 32 + mm] = vb_[gk];
        }
#endif
        __syncthreads();

        // S(16x32) = Q(16x64) * K^T : B[k=d][n=m] = Ks[m][d]
        f32x8 S0 = {}, S1 = {};
        S0 = wmma_bf16(qa0, load_b_frag(Ks, 64, 0, 0, lane), S0);
        S0 = wmma_bf16(qa1, load_b_frag(Ks, 64, 0, 32, lane), S0);
        S1 = wmma_bf16(qa0, load_b_frag(Ks, 64, 16, 0, lane), S1);
        S1 = wmma_bf16(qa1, load_b_frag(Ks, 64, 16, 32, lane), S1);

        unsigned short* Pw = Ps[wave];
#pragma unroll
        for (int r = 0; r < 8; ++r) {
            float s0 = S0[r] * scale, s1 = S1[r] * scale;
            float mx = fmaxf(s0, s1);
#pragma unroll
            for (int off = 8; off > 0; off >>= 1)
                mx = fmaxf(mx, __shfl_xor(mx, off, 32));
            float mnew  = fmaxf(m_run[r], mx);
            float alpha = __expf(m_run[r] - mnew);
            float p0 = __expf(s0 - mnew), p1 = __expf(s1 - mnew);
            float rs = p0 + p1;
#pragma unroll
            for (int off = 8; off > 0; off >>= 1)
                rs += __shfl_xor(rs, off, 32);
            l_run[r] = l_run[r] * alpha + rs;
            m_run[r] = mnew;
            O[0][r] *= alpha; O[1][r] *= alpha;
            O[2][r] *= alpha; O[3][r] *= alpha;
            // C-layout -> row-major scratch (row = mOff+r, col = 0..31)
            Pw[(mOff + r) * 32 + nloc]      = f32_to_bf16(p0);
            Pw[(mOff + r) * 32 + 16 + nloc] = f32_to_bf16(p1);
        }
        // Same-wave DS RAW: LDS is in-order per wave; wait + compiler fence.
        asm volatile("s_wait_dscnt 0" ::: "memory");
        bf16x16 pa = load_a_frag(Pw, 32, 0, 0, lane);
        // O(16x64) += P(16x32) * V(32x64): B[k=m][n=d] = Vt[d][m]
#pragma unroll
        for (int j = 0; j < 4; ++j)
            O[j] = wmma_bf16(pa, load_b_frag(Vt, 32, j * 16, 0, lane), O[j]);
    }

    // normalize and store [B,L,H*64] bf16
#pragma unroll
    for (int r = 0; r < 8; ++r) {
        float invl = 1.0f / l_run[r];
        int rowo = qRow0 + mOff + r;
        size_t obase = ((size_t)b * L + rowo) * (size_t)(H * 64) + (size_t)h * 64;
#pragma unroll
        for (int j = 0; j < 4; ++j)
            out[obase + j * 16 + nloc] = f32_to_bf16(O[j][r] * invl);
    }
}

// ---------------------------------------------------------------------------
// Launcher. Workspace map (needs >= 32 MB):
//   [0,16M)   q_lin f32 (4096x1024)        -- reused for attn_bf (8 MB) later
//   [16,20M)  k_lin f32 (4096x256)
//   [20,28M)  q_bf  bf16 (4096x1024)
//   [28,30M)  k_bf  bf16 (4096x256)
//   [30,32M)  v_bf  bf16 (4096x256)
// ---------------------------------------------------------------------------
extern "C" void kernel_launch(void* const* d_in, const int* in_sizes, int n_in,
                              void* d_out, int out_size, void* d_ws, size_t ws_size,
                              hipStream_t stream) {
    (void)in_sizes; (void)n_in; (void)out_size; (void)ws_size;
    const float* query = (const float*)d_in[0];
    const float* key   = (const float*)d_in[1];
    const float* value = (const float*)d_in[2];
    const float* Wq = (const float*)d_in[3];
    const float* bq = (const float*)d_in[4];
    const float* Wk = (const float*)d_in[5];
    const float* bk = (const float*)d_in[6];
    const float* Wv = (const float*)d_in[7];
    const float* bv = (const float*)d_in[8];
    const float* Wo = (const float*)d_in[9];
    const float* bo = (const float*)d_in[10];
    float* out = (float*)d_out;

    const int B = 4, L = 1024, D = 1024, H = 16, G = 4;
    const int M = B * L;          // 4096
    const int KV = G * 64;        // 256

    char* ws = (char*)d_ws;
    float*          q_lin   = (float*)(ws);
    float*          k_lin   = (float*)(ws + ((size_t)16 << 20));
    unsigned short* q_bf    = (unsigned short*)(ws + ((size_t)20 << 20));
    unsigned short* k_bf    = (unsigned short*)(ws + ((size_t)28 << 20));
    unsigned short* v_bf    = (unsigned short*)(ws + ((size_t)30 << 20));
    unsigned short* attn_bf = (unsigned short*)(ws);   // reuse q_lin region

    dim3 blk(128);
    // Projections
    gemm_bt_kernel<false, false><<<dim3(M / 64, D / 64), blk, 0, stream>>>(
        query, Wq, bq, q_lin, M, D, D);
    gemm_bt_kernel<false, false><<<dim3(M / 64, KV / 64), blk, 0, stream>>>(
        key, Wk, bk, k_lin, M, KV, D);
    gemm_bt_kernel<false, true><<<dim3(M / 64, KV / 64), blk, 0, stream>>>(
        value, Wv, bv, v_bf, M, KV, D);
    // RoPE (q,k) -> bf16
    int totq = M * H * 32;
    rope_kernel<<<(totq + 255) / 256, 256, 0, stream>>>(q_lin, q_bf, H, L, totq);
    int totk = M * G * 32;
    rope_kernel<<<(totk + 255) / 256, 256, 0, stream>>>(k_lin, k_bf, G, L, totk);
    // Attention (writes over q_lin region, which is dead after RoPE)
    attn_kernel<<<dim3(B * H * (L / 64)), blk, 0, stream>>>(
        q_bf, k_bf, v_bf, attn_bf, B, L, H, G);
    // Output projection -> f32 d_out
    gemm_bt_kernel<true, false><<<dim3(M / 64, D / 64), blk, 0, stream>>>(
        attn_bf, Wo, bo, out, M, D, D);
}
